// GearNetLayer_37220186587485
// MI455X (gfx1250) — matrix operand order repaired and checked by reading, not compiled, not run on hardware
//
#include <hip/hip_runtime.h>
#include <hip/hip_bf16.h>

typedef _Float16 v16h __attribute__((ext_vector_type(16)));
typedef _Float16 v8h  __attribute__((ext_vector_type(8)));
typedef float    v8f  __attribute__((ext_vector_type(8)));

#define DIM       128
#define TILE_E    64    // edges per block iteration (4 x 16 M-subtiles)
#define MT        4     // M-subtiles per tile
#define THREADS   256
#define XH_STRIDE 136   // halfs per row, padded (272B: 16B aligned, breaks 64-bank stride)

// ---------------------------------------------------------------------------
// Kernel 1: fold the rank-1 edge_attr path into two 128-vectors:
//   u[j] = sum_k W_edge[0,k] * W1[D+k, j]
//   c[j] = sum_k b_edge[k]   * W1[D+k, j] + b1[j]
// ---------------------------------------------------------------------------
__global__ void prep_uc_kernel(const float* __restrict__ W_edge,
                               const float* __restrict__ b_edge,
                               const float* __restrict__ W1,
                               const float* __restrict__ b1,
                               float* __restrict__ uc) {
  int j = threadIdx.x;
  if (j < DIM) {
    float u = 0.f, c = 0.f;
    for (int k = 0; k < DIM; ++k) {
      float w = W1[(DIM + k) * DIM + j];
      u += W_edge[k] * w;
      c += b_edge[k] * w;
    }
    uc[j]       = u;
    uc[DIM + j] = c + b1[j];
  }
}

// ---------------------------------------------------------------------------
// Kernel 2: out = x  (residual; edge kernel atomically accumulates into it)
// ---------------------------------------------------------------------------
__global__ void copy_x_kernel(const float4* __restrict__ x,
                              float4* __restrict__ out, int n4) {
  int i = blockIdx.x * blockDim.x + threadIdx.x;
  if (i < n4) out[i] = x[i];
}

// ---------------------------------------------------------------------------
// Kernel 3: fused gather -> GEMM1(+rank1+ReLU) -> GEMM2 -> atomic scatter
// 256 threads = 8 waves; wave w owns output columns [16w, 16w+16).
// Weights live in per-wave registers (B-panels); 64-edge tiles amortize
// barriers/scatter over 32 WMMAs per wave per tile.
// ---------------------------------------------------------------------------
__global__ void __launch_bounds__(THREADS)
edge_mlp_scatter_kernel(const float* __restrict__ x,
                        const float* __restrict__ coord,
                        const long long* __restrict__ eidx,   // [2][E]
                        const float* __restrict__ W1,
                        const float* __restrict__ W2,
                        const float* __restrict__ b2,
                        const float* __restrict__ uc,
                        float* __restrict__ out,
                        int nTiles, long long E) {
  __shared__ __align__(16) _Float16 lds_x[TILE_E * XH_STRIDE];
  __shared__ __align__(16) _Float16 lds_h[TILE_E * XH_STRIDE];
  __shared__ int   s_row[TILE_E];
  __shared__ int   s_col[TILE_E];
  __shared__ float s_dist[TILE_E];

  const int tid  = threadIdx.x;
  const int lane = tid & 31;
  const int wave = tid >> 5;          // n-tile id 0..7
  const int nlo  = lane & 15;         // A: row M ; B/C/D: column within tile
  const int hi   = lane >> 4;         // lane half
  const int ng   = wave * 16 + nlo;   // global output column 0..127
  const int kh8  = hi * 8;            // A-matrix K sub-offset
  const int kh16 = hi * 16;           // unused; B panels also use kh8

  // --- Per-wave register-resident B panels (same K order as A:
  //     lanes 0-15 hold K=k0..k0+7,k0+16..k0+23; lanes 16-31 hold K=k0+8..k0+15,k0+24..k0+31).
  v16h bw1[4], bw2[4];
#pragma unroll
  for (int ks = 0; ks < 4; ++ks) {
    const int k0 = ks * 32 + kh8;
#pragma unroll
    for (int i = 0; i < 16; ++i) {
      bw1[ks][i] = (_Float16)W1[(k0 + i + (i & 8)) * DIM + ng];   // top half of W1
      bw2[ks][i] = (_Float16)W2[(k0 + i + (i & 8)) * DIM + ng];
    }
  }
  const float u_l  = uc[ng];
  const float c_l  = uc[DIM + ng];
  const float b2_l = b2[ng];

  for (int tile = blockIdx.x; tile < nTiles; tile += gridDim.x) {
    const long long ebase = (long long)tile * TILE_E;

    // --- edge indices + pairwise distance (64 edges) -----------------------
    if (tid < TILE_E) {
      long long e = ebase + tid;
      int r = 0, c = 0;
      float dist = 0.f;
      if (e < E) {
        r = (int)eidx[e];
        c = (int)eidx[E + e];
        float dx = coord[3 * r + 0] - coord[3 * c + 0];
        float dy = coord[3 * r + 1] - coord[3 * c + 1];
        float dz = coord[3 * r + 2] - coord[3 * c + 2];
        dist = sqrtf(dx * dx + dy * dy + dz * dz);
      }
      s_row[tid] = r; s_col[tid] = c; s_dist[tid] = dist;

      // prefetch next tile's indices into cache (global_prefetch_b8)
      long long en = e + (long long)gridDim.x * TILE_E;
      if (en < E) {
        __builtin_prefetch(&eidx[en], 0, 0);
        __builtin_prefetch(&eidx[E + en], 0, 0);
      }
    }
    __syncthreads();

    // --- gather x[col] -> LDS, f32 -> f16 (64 rows x 128) -----------------
#pragma unroll
    for (int mb = 0; mb < MT; ++mb) {
      const int m  = mb * 16 + (tid >> 4);   // 0..63 edge row
      const int kb = (tid & 15) * 8;         // 0..120
      const float* src = x + (size_t)s_col[m] * DIM + kb;
      float4 a = *(const float4*)(src);
      float4 b = *(const float4*)(src + 4);
      _Float16* dst = &lds_x[m * XH_STRIDE + kb];
      dst[0] = (_Float16)a.x; dst[1] = (_Float16)a.y;
      dst[2] = (_Float16)a.z; dst[3] = (_Float16)a.w;
      dst[4] = (_Float16)b.x; dst[5] = (_Float16)b.y;
      dst[6] = (_Float16)b.z; dst[7] = (_Float16)b.w;
    }
    __syncthreads();

    // --- GEMM1: h = relu(x_col @ W1_top + dist*u + c), 4 M-subtiles -------
#pragma unroll
    for (int mt = 0; mt < MT; ++mt) {
      const int mrow = mt * 16 + nlo;
      v8f acc = {};
#pragma unroll
      for (int ks = 0; ks < 4; ++ks) {
        const int k0 = ks * 32;
        v8h alo = *(const v8h*)&lds_x[mrow * XH_STRIDE + k0 + kh8];
        v8h ahi = *(const v8h*)&lds_x[mrow * XH_STRIDE + k0 + 16 + kh8];
        v16h a = __builtin_shufflevector(alo, ahi,
            0, 1, 2, 3, 4, 5, 6, 7, 8, 9, 10, 11, 12, 13, 14, 15);
        acc = __builtin_amdgcn_wmma_f32_16x16x32_f16(
            false, a, false, bw1[ks], (short)0, acc, false, false);
      }
      // C/D layout: VGPR r -> M = r + 8*hi (within subtile), column = ng.
#pragma unroll
      for (int r = 0; r < 8; ++r) {
        const int m = mt * 16 + r + hi * 8;
        float v = acc[r] + s_dist[m] * u_l + c_l;
        v = v > 0.f ? v : 0.f;
        lds_h[m * XH_STRIDE + ng] = (_Float16)v;   // implicit transpose via LDS
      }
    }
    __syncthreads();

    // --- GEMM2: msg = h @ W2 + b2; atomic scatter into out[row] -----------
#pragma unroll
    for (int mt = 0; mt < MT; ++mt) {
      const int mrow = mt * 16 + nlo;
      v8f acc2 = {};
#pragma unroll
      for (int ks = 0; ks < 4; ++ks) {
        const int k0 = ks * 32;
        v8h alo = *(const v8h*)&lds_h[mrow * XH_STRIDE + k0 + kh8];
        v8h ahi = *(const v8h*)&lds_h[mrow * XH_STRIDE + k0 + 16 + kh8];
        v16h a = __builtin_shufflevector(alo, ahi,
            0, 1, 2, 3, 4, 5, 6, 7, 8, 9, 10, 11, 12, 13, 14, 15);
        acc2 = __builtin_amdgcn_wmma_f32_16x16x32_f16(
            false, a, false, bw2[ks], (short)0, acc2, false, false);
      }
#pragma unroll
      for (int r = 0; r < 8; ++r) {
        const int m = mt * 16 + r + hi * 8;
        if (ebase + m < E) {
          unsafeAtomicAdd(&out[(size_t)s_row[m] * DIM + ng], acc2[r] + b2_l);
        }
      }
    }
    __syncthreads();
  }
}

// ---------------------------------------------------------------------------
extern "C" void kernel_launch(void* const* d_in, const int* in_sizes, int n_in,
                              void* d_out, int out_size, void* d_ws, size_t ws_size,
                              hipStream_t stream) {
  const float*     x      = (const float*)d_in[0];
  const float*     coord  = (const float*)d_in[1];
  const long long* eidx   = (const long long*)d_in[2];
  const float*     W_edge = (const float*)d_in[3];
  const float*     b_edge = (const float*)d_in[4];
  const float*     W1     = (const float*)d_in[5];
  const float*     b1     = (const float*)d_in[6];
  const float*     W2     = (const float*)d_in[7];
  const float*     b2     = (const float*)d_in[8];
  float*           out    = (float*)d_out;
  float*           uc     = (float*)d_ws;   // 256 floats

  const long long E = (long long)(in_sizes[2] / 2);
  const int nTiles  = (int)((E + TILE_E - 1) / TILE_E);

  // 1) rank-1 fold of the edge_attr path
  prep_uc_kernel<<<1, DIM, 0, stream>>>(W_edge, b_edge, W1, b1, uc);

  // 2) residual init: out = x
  const int n4 = in_sizes[0] / 4;
  copy_x_kernel<<<(n4 + THREADS - 1) / THREADS, THREADS, 0, stream>>>(
      (const float4*)x, (float4*)out, n4);

  // 3) fused edge MLP + scatter (persistent blocks)
  int nBlocks = nTiles < 4096 ? nTiles : 4096;
  edge_mlp_scatter_kernel<<<nBlocks, THREADS, 0, stream>>>(
      x, coord, eidx, W1, W2, b2, uc, out, nTiles, E);
}
